// TripleAttentionSmarter_21809843929814
// MI455X (gfx1250) — matrix operand order repaired and checked by reading, not compiled, not run on hardware
//
#include <hip/hip_runtime.h>

// ---------------------------------------------------------------------------
// TripleAttention for gfx1250 (MI455X): bf16 WMMA pipeline.
// Sizes: B=2, N=256, C=512, H=8, D=64 -> BH=16, BN=512.
// Dominant cost: diag einsum O(BH*D*N^3) ~ 34 GFLOP; all data L2-resident
// (192MB L2), so everything runs through v_wmma_f32_16x16x32_bf16.
// Wave tile is 64x32 (4x2 accumulators = 64 VGPRs) to stay spill-free.
// ---------------------------------------------------------------------------

#define DEV __device__ __forceinline__

typedef __bf16 bf16;
typedef __bf16 v16bf __attribute__((ext_vector_type(16)));
typedef __bf16 v8bf  __attribute__((ext_vector_type(8)));
typedef float  v8f   __attribute__((ext_vector_type(8)));

#define B_    2
#define N_    256
#define C_    512
#define H_    8
#define D_    64
#define BH_   16
#define BN_   512
#define NN_   (N_ * N_)      /* 65536  */
#define ND_   (N_ * D_)      /* 16384  */
#define BHND_ (BH_ * ND_)    /* 262144 */
#define SCALE_ 0.125f        /* D^-0.5 */

DEV v8f zero8() {
  v8f z;
#pragma unroll
  for (int e = 0; e < 8; ++e) z[e] = 0.0f;
  return z;
}

// A-fragment: 16x32 bf16, A row-major [M][K], stride ld (elements, mult of 8).
// lane m = lane&15, half = lane>>4; VGPR v<4: K = half*8 + 2v(+1); v>=4: +16.
// => two contiguous 16B loads at k0+half*8 and k0+16+half*8.
DEV v16bf load_fragA(const bf16* A, int row0, int k0, int ld) {
  const int lane = threadIdx.x & 31;
  const int m = lane & 15, hf = lane >> 4;
  const bf16* p = A + (row0 + m) * ld + k0 + hf * 8;
  v8bf lo = *(const v8bf*)p;
  v8bf hi = *(const v8bf*)(p + 16);
  v16bf r;
#pragma unroll
  for (int e = 0; e < 8; ++e) { r[e] = lo[e]; r[8 + e] = hi[e]; }
  return r;
}

// B-fragment: 32x16 bf16 of B[k][n], supplied as Bt row-major [N][K]
// (i.e. computes A @ Bt^T). lane n = lane&15, half = lane>>4;
// VGPR v<4: K = half*8 + 2v(+1); v>=4: +16 => two 16B loads (same as A).
DEV v16bf load_fragB(const bf16* Bt, int n0, int k0, int ld) {
  const int lane = threadIdx.x & 31;
  const int n = lane & 15, hf = lane >> 4;
  const bf16* p = Bt + (n0 + n) * ld + k0 + hf * 8;
  v8bf lo = *(const v8bf*)p;
  v8bf hi = *(const v8bf*)(p + 16);
  v16bf r;
#pragma unroll
  for (int e = 0; e < 8; ++e) { r[e] = lo[e]; r[8 + e] = hi[e]; }
  return r;
}

// One wave computes a 64x32 f32 tile of A(row-major) @ Bt^T(row-major).
// 4x2 accumulators (64 VGPRs) + 6 fragment registers (48 VGPRs): spill-free.
DEV void gemm64x32(const bf16* A, const bf16* Bt, int row0, int col0,
                   int lda, int ldb, int K, v8f acc[4][2]) {
#pragma unroll
  for (int si = 0; si < 4; ++si)
#pragma unroll
    for (int sj = 0; sj < 2; ++sj) acc[si][sj] = zero8();
  for (int k0 = 0; k0 < K; k0 += 32) {
    v16bf a[4], bt[2];
#pragma unroll
    for (int s = 0; s < 4; ++s) a[s] = load_fragA(A, row0 + 16 * s, k0, lda);
#pragma unroll
    for (int s = 0; s < 2; ++s) bt[s] = load_fragB(Bt, col0 + 16 * s, k0, ldb);
#pragma unroll
    for (int si = 0; si < 4; ++si)
#pragma unroll
      for (int sj = 0; sj < 2; ++sj)
        acc[si][sj] = __builtin_amdgcn_wmma_f32_16x16x32_bf16(
            false, a[si], false, bt[sj], (short)0, acc[si][sj], false, false);
  }
}

// ---------------------------------------------------------------------------
__global__ void cvt_kernel(const float* __restrict__ in, bf16* __restrict__ out, int n) {
  int i = blockIdx.x * 256 + threadIdx.x;
  if (i < n) out[i] = (bf16)in[i];
}

// w = x @ Ww^T, scattered head-major: w5[q][bh][n][d], q in 0..4 (a,b,c,v1,v2).
__global__ __launch_bounds__(32, 1)
void proj_kernel(const bf16* __restrict__ xb, const bf16* __restrict__ Wwb,
                 bf16* __restrict__ w5) {
  const int tm = blockIdx.x * 64;   // rows of (BN x 5C) output
  const int tn = blockIdx.y * 32;   // cols (32-aligned -> fixed q,h per tile)
  v8f acc[4][2];
  gemm64x32(xb, Wwb, tm, tn, C_, C_, C_, acc);
  const int lane = threadIdx.x & 31, n = lane & 15, hf = lane >> 4;
  const int q = tn >> 9;
  const int h = (tn & 511) >> 6;
  const int dbase = tn & 63;        // 0 or 32 within the 64-wide head block
#pragma unroll
  for (int si = 0; si < 4; ++si)
#pragma unroll
    for (int sj = 0; sj < 2; ++sj)
#pragma unroll
      for (int v = 0; v < 8; ++v) {
        const int row = tm + si * 16 + v + 8 * hf;   // row in BN
        const int bidx = row >> 8, tok = row & 255;
        const int dcol = dbase + sj * 16 + n;        // 0..63 within head
        w5[q * BHND_ + (bidx * H_ + h) * ND_ + tok * D_ + dcol] =
            (bf16)acc[si][sj][v];
      }
}

// P_p = exp(scale * Aq @ Bq^T), pairs (a,b),(b,c),(c,a). Stores P and P^T.
__global__ __launch_bounds__(32, 1)
void scores_kernel(const bf16* __restrict__ w5, bf16* __restrict__ XYZ,
                   bf16* __restrict__ XYZt) {
  const int t = blockIdx.x, bh = blockIdx.y, p = blockIdx.z;
  const int tm = (t >> 3) * 64, tn = (t & 7) * 32;
  const int qa = p, qb = (p == 2) ? 0 : p + 1;
  v8f acc[4][2];
  gemm64x32(w5 + qa * BHND_ + bh * ND_, w5 + qb * BHND_ + bh * ND_,
            tm, tn, D_, D_, D_, acc);
  bf16* P  = XYZ  + (size_t)(p * BH_ + bh) * NN_;
  bf16* Pt = XYZt + (size_t)(p * BH_ + bh) * NN_;
  const int lane = threadIdx.x & 31, n = lane & 15, hf = lane >> 4;
#pragma unroll
  for (int si = 0; si < 4; ++si)
#pragma unroll
    for (int sj = 0; sj < 2; ++sj)
#pragma unroll
      for (int v = 0; v < 8; ++v) {
        const int i = tm + si * 16 + v + 8 * hf;
        const int j = tn + sj * 16 + n;
        const float e = __expf(SCALE_ * acc[si][sj][v]);
        P[i * N_ + j] = (bf16)e;
        Pt[j * N_ + i] = (bf16)e;
      }
}

// XY = X @ Y (f32 out), via Yt so B-fragments are contiguous.
__global__ __launch_bounds__(32, 1)
void xy_kernel(const bf16* __restrict__ Xb, const bf16* __restrict__ Ytb,
               float* __restrict__ XYf) {
  const int t = blockIdx.x, bh = blockIdx.y;
  const int tm = (t >> 3) * 64, tn = (t & 7) * 32;
  v8f acc[4][2];
  gemm64x32(Xb + (size_t)bh * NN_, Ytb + (size_t)bh * NN_, tm, tn, N_, N_, N_, acc);
  const int lane = threadIdx.x & 31, n = lane & 15, hf = lane >> 4;
  float* o = XYf + (size_t)bh * NN_;
#pragma unroll
  for (int si = 0; si < 4; ++si)
#pragma unroll
    for (int sj = 0; sj < 2; ++sj)
#pragma unroll
      for (int v = 0; v < 8; ++v)
        o[(tm + si * 16 + v + 8 * hf) * N_ + tn + sj * 16 + n] = acc[si][sj][v];
}

// den[bh][i] = sum_k XY[i,k] * Z[k,i]
__global__ void den_kernel(const float* __restrict__ XYf, const bf16* __restrict__ Zb,
                           float* __restrict__ den) {
  const int bh = blockIdx.x, i = threadIdx.x;
  const float* xy = XYf + (size_t)bh * NN_ + i * N_;
  const bf16* z = Zb + (size_t)bh * NN_;
  float s = 0.0f;
  for (int k = 0; k < N_; ++k) s += xy[k] * (float)z[k * N_ + i];
  den[bh * N_ + i] = s;
}

// num[bh][i][d] = sum_k ( sum_j (X[i,j]*v1[j,d]) * Y[j,k] ) * Z[k,i] * v2[k,d]
// One wave per (i-block of 64, d, bh). Xu staged/scaled in LDS once, reused
// across 8 k-blocks of 32; reduction over k via wave32 shfl_xor butterflies.
__global__ __launch_bounds__(32, 1)
void contract_kernel(const bf16* __restrict__ Xb, const bf16* __restrict__ Ytb,
                     const bf16* __restrict__ Zb, const bf16* __restrict__ v1b,
                     const bf16* __restrict__ v2b, float* __restrict__ num) {
  __shared__ __align__(16) bf16 XuL[64][N_];   // 32 KB
  __shared__ float v1col[N_];
  const int lane = threadIdx.x & 31;
  const int i0 = blockIdx.x * 64;
  const int d  = blockIdx.y;
  const int bh = blockIdx.z;
  const bf16* Xr = Xb  + (size_t)bh * NN_;
  const bf16* Yt = Ytb + (size_t)bh * NN_;
  const bf16* Zr = Zb  + (size_t)bh * NN_;

  for (int j = lane; j < N_; j += 32) v1col[j] = (float)v1b[bh * ND_ + j * D_ + d];
  __syncthreads();
  for (int idx = lane; idx < 64 * N_; idx += 32) {
    const int r = idx >> 8, j = idx & (N_ - 1);
    XuL[r][j] = (bf16)((float)Xr[(i0 + r) * N_ + j] * v1col[j]);
  }
  __syncthreads();

  const int n = lane & 15, hf = lane >> 4;
  float nacc[4][8];
#pragma unroll
  for (int si = 0; si < 4; ++si)
#pragma unroll
    for (int v = 0; v < 8; ++v) nacc[si][v] = 0.0f;

  const bf16* XuP = &XuL[0][0];
  for (int kb = 0; kb < 8; ++kb) {
    const int k0 = kb * 32;
    __builtin_prefetch(Zr + k0 * N_ + i0, 0, 1);   // global_prefetch for Z tile
    v8f acc[4][2];
#pragma unroll
    for (int si = 0; si < 4; ++si)
#pragma unroll
      for (int sj = 0; sj < 2; ++sj) acc[si][sj] = zero8();
    for (int j0 = 0; j0 < N_; j0 += 32) {
      v16bf a[4], bt[2];
#pragma unroll
      for (int s = 0; s < 4; ++s) a[s] = load_fragA(XuP, 16 * s, j0, N_);
#pragma unroll
      for (int s = 0; s < 2; ++s) bt[s] = load_fragB(Yt, k0 + 16 * s, j0, N_);
#pragma unroll
      for (int si = 0; si < 4; ++si)
#pragma unroll
        for (int sk = 0; sk < 2; ++sk)
          acc[si][sk] = __builtin_amdgcn_wmma_f32_16x16x32_bf16(
              false, a[si], false, bt[sk], (short)0, acc[si][sk], false, false);
    }
    // reduce: acc[i,k] * Z[k,i] * v2[k,d], summed over k (lanes + sk)
#pragma unroll
    for (int sk = 0; sk < 2; ++sk) {
      const int k = k0 + sk * 16 + n;
      const float v2k = (float)v2b[bh * ND_ + k * D_ + d];
#pragma unroll
      for (int si = 0; si < 4; ++si)
#pragma unroll
        for (int v = 0; v < 8; ++v) {
          const int i = i0 + si * 16 + v + 8 * hf;
          float t = acc[si][sk][v] * (float)Zr[k * N_ + i] * v2k;
          t += __shfl_xor(t, 1);
          t += __shfl_xor(t, 2);
          t += __shfl_xor(t, 4);
          t += __shfl_xor(t, 8);   // stays within 16-lane half
          nacc[si][v] += t;
        }
    }
  }
  if (n == 0) {
#pragma unroll
    for (int si = 0; si < 4; ++si)
#pragma unroll
      for (int v = 0; v < 8; ++v) {
        const int i = i0 + si * 16 + v + 8 * hf;
        num[bh * ND_ + i * D_ + d] = nacc[si][v];
      }
  }
}

__global__ void finalize_kernel(const float* __restrict__ num, const float* __restrict__ den,
                                bf16* __restrict__ Hb) {
  const int idx = blockIdx.x * 256 + threadIdx.x;  // < BHND_
  const int bh = idx >> 14, i = (idx >> 6) & 255;
  Hb[idx] = (bf16)(num[idx] / den[bh * N_ + i]);
}

// out = H @ Wo^T + bo   (H is the (B,H,N,D) buffer viewed as (BN x C) rows)
__global__ __launch_bounds__(32, 1)
void out_gemm(const bf16* __restrict__ Hb, const bf16* __restrict__ Wob,
              const float* __restrict__ bo, float* __restrict__ out) {
  const int tm = blockIdx.x * 64, tn = blockIdx.y * 32;
  v8f acc[4][2];
  gemm64x32(Hb, Wob, tm, tn, C_, C_, C_, acc);
  const int lane = threadIdx.x & 31, n = lane & 15, hf = lane >> 4;
#pragma unroll
  for (int si = 0; si < 4; ++si)
#pragma unroll
    for (int sj = 0; sj < 2; ++sj)
#pragma unroll
      for (int v = 0; v < 8; ++v) {
        const int row = tm + si * 16 + v + 8 * hf;
        const int col = tn + sj * 16 + n;
        out[row * C_ + col] = acc[si][sj][v] + bo[col];
      }
}

// ---------------------------------------------------------------------------
extern "C" void kernel_launch(void* const* d_in, const int* in_sizes, int n_in,
                              void* d_out, int out_size, void* d_ws, size_t ws_size,
                              hipStream_t stream) {
  const float* x  = (const float*)d_in[0];   // (B,N,C)
  const float* Ww = (const float*)d_in[1];   // (5C,C)
  const float* Wo = (const float*)d_in[2];   // (C,C)
  const float* bo = (const float*)d_in[3];   // (C,)
  float* out = (float*)d_out;

  char* ws = (char*)d_ws;
  size_t off = 0;
  bf16* xb   = (bf16*)(ws + off); off += (size_t)BN_ * C_ * 2;       // 512 KB
  bf16* Wwb  = (bf16*)(ws + off); off += (size_t)5 * C_ * C_ * 2;    // 2.5 MB
  bf16* Wob  = (bf16*)(ws + off); off += (size_t)C_ * C_ * 2;        // 512 KB
  bf16* w5b  = (bf16*)(ws + off); off += (size_t)5 * BHND_ * 2;      // 2.5 MB
  bf16* XYZ  = (bf16*)(ws + off); off += (size_t)3 * BH_ * NN_ * 2;  // 6 MB
  bf16* XYZt = (bf16*)(ws + off); off += (size_t)3 * BH_ * NN_ * 2;  // 6 MB
  float* XYf = (float*)(ws + off); off += (size_t)BH_ * NN_ * 4;     // 4 MB
  float* denf= (float*)(ws + off); off += (size_t)BH_ * N_ * 4;      // 16 KB
  float* numf= (float*)(ws + off); off += (size_t)BHND_ * 4;         // 1 MB
  bf16* Hb   = (bf16*)(ws + off); off += (size_t)BHND_ * 2;          // 512 KB
  (void)ws_size; (void)in_sizes; (void)n_in; (void)out_size;         // ~23.6 MB total

  cvt_kernel<<<(BN_ * C_ + 255) / 256, 256, 0, stream>>>(x, xb, BN_ * C_);
  cvt_kernel<<<(5 * C_ * C_ + 255) / 256, 256, 0, stream>>>(Ww, Wwb, 5 * C_ * C_);
  cvt_kernel<<<(C_ * C_ + 255) / 256, 256, 0, stream>>>(Wo, Wob, C_ * C_);

  proj_kernel<<<dim3(BN_ / 64, (5 * C_) / 32), 32, 0, stream>>>(xb, Wwb, w5b);
  scores_kernel<<<dim3(32, BH_, 3), 32, 0, stream>>>(w5b, XYZ, XYZt);

  const bf16* Ytb = XYZt + (size_t)BH_ * NN_;        // p=1 is Y
  const bf16* Zb  = XYZ  + (size_t)2 * BH_ * NN_;    // p=2 is Z
  xy_kernel<<<dim3(32, BH_), 32, 0, stream>>>(XYZ, Ytb, XYf);
  den_kernel<<<BH_, N_, 0, stream>>>(XYf, Zb, denf);

  contract_kernel<<<dim3(N_ / 64, D_, BH_), 32, 0, stream>>>(
      XYZ, Ytb, Zb, w5b + (size_t)3 * BHND_, w5b + (size_t)4 * BHND_, numf);

  finalize_kernel<<<(BHND_ + 255) / 256, 256, 0, stream>>>(numf, denf, Hb);
  out_gemm<<<dim3(BN_ / 64, C_ / 32), 32, 0, stream>>>(Hb, Wob, bo, out);
}
